// Modulated_FC_Conv_34445637714486
// MI455X (gfx1250) — matrix operand order repaired and checked by reading, not compiled, not run on hardware
//
#include <hip/hip_runtime.h>

// ---------------------------------------------------------------------------
// Modulated 1x1 conv (StyleGAN2 style):
//   s = style @ (mod_w*MOD_SCALE)^T + mod_b + 1          (4, 256)
//   w = SCALE * weight * s ; demod over in_c ; w *= demod (4, 256, 256)
//   out[b] = x[b] (65536x256) @ w[b] (256x256)            (4, 65536, 256)
//
// Weight prep kernel writes w in CDNA5 WMMA B-fragment layout (bf16).
// Main kernel DMAs the 128KB panel into LDS with the Tensor Data Mover
// (tensor_load_to_lds, TENSORcnt), then runs v_wmma_f32_16x16x32_bf16
// with fp32 accumulation. Memory-bound by design (512MB x/out traffic
// @ 23.3 TB/s ~ 22us floor).
// ---------------------------------------------------------------------------

typedef __attribute__((ext_vector_type(16))) __bf16 v16bf;
typedef __attribute__((ext_vector_type(8)))  float  v8f;
typedef __attribute__((ext_vector_type(4)))  unsigned int v4u;
typedef __attribute__((ext_vector_type(8)))  int    v8i;
typedef __attribute__((ext_vector_type(4)))  int    v4i;

#define IN_C       256
#define OUT_C      256
#define STYLE_DIM  512
#define N_ROWS     65536
#define BATCH      4
#define SCALE_     0.0625f               /* 1/sqrt(256) */
#define MOD_SCALE_ 0.044194173824159216f /* 1/sqrt(512) */
#define EPS_       1e-8f

// w-fragment layout per batch: [kt(8)][nt(16)][lane(32)][i(16)] bf16
//   lane <16 : col o = nt*16 + lane,      K = kt*32 + i        (i=0..15)
//   lane>=16 : col o = nt*16 + lane-16,   K = kt*32 + 16 + i
// 8*16*32*16 = 65536 bf16 = 128KB per batch; 512KB total in d_ws.

__global__ __launch_bounds__(256) void prep_weights_kernel(
    const float* __restrict__ style,   // (4, 512)
    const float* __restrict__ weight,  // (256, 256) = (in, out)
    const float* __restrict__ mod_w,   // (256, 512)
    const float* __restrict__ mod_b,   // (256)
    __bf16* __restrict__ wfrag)        // (4, 65536) bf16
{
    __shared__ float s_sh[IN_C];
    __shared__ float demod_sh[OUT_C];
    const int b   = blockIdx.x;
    const int tid = threadIdx.x;

    // s[c] for c = tid
    {
        const float* st = style + b * STYLE_DIM;
        const float* mw = mod_w + tid * STYLE_DIM;
        float acc = 0.f;
        for (int j = 0; j < STYLE_DIM; ++j) acc += st[j] * mw[j];
        s_sh[tid] = acc * MOD_SCALE_ + mod_b[tid] + 1.0f;
    }
    __syncthreads();

    // demod[o] for o = tid
    {
        float ss = 0.f;
        for (int c = 0; c < IN_C; ++c) {
            float w = SCALE_ * weight[c * OUT_C + tid] * s_sh[c];
            ss += w * w;
        }
        demod_sh[tid] = rsqrtf(ss + EPS_);
    }
    __syncthreads();

    // Emit B-fragment layout
    for (int slot = tid; slot < 8 * 16 * 32; slot += 256) {
        const int kt   = slot >> 9;        // / (16*32)
        const int rem  = slot & 511;
        const int nt   = rem >> 5;
        const int lane = rem & 31;
        const int o     = nt * 16 + (lane & 15);
        const int cBase = kt * 32 + (lane >> 4) * 16;
        const float dm  = demod_sh[o];
        __bf16* dst = wfrag + ((size_t)b << 16) + (size_t)slot * 16;
        #pragma unroll
        for (int i = 0; i < 16; ++i) {
            const int c = cBase + i;
            dst[i] = (__bf16)(SCALE_ * weight[c * OUT_C + o] * s_sh[c] * dm);
        }
    }
}

// Each block: one batch, 128 rows of x. 8 waves, 16 rows per wave.
// LDS: full 128KB B panel for the batch (dynamic shared), filled by TDM.
__global__ __launch_bounds__(256) void modconv_gemm_kernel(
    const float* __restrict__ x,       // (4, 65536, 256)
    const __bf16* __restrict__ wfrag,  // (4, 65536) bf16
    float* __restrict__ out)           // (4, 65536, 256)
{
    extern __shared__ char smem_raw[];
    __bf16* smemB = (__bf16*)smem_raw;          // 65536 bf16 = 128KB

    const int tid  = threadIdx.x;
    const int b    = blockIdx.x >> 9;           // 512 blocks per batch
    const int mblk = blockIdx.x & 511;

    // --- Stage B panel into LDS with the Tensor Data Mover --------------
    // One TDM descriptor: copy 16384 x 8B (= 128KB, contiguous) from
    // wfrag[b] to LDS offset 0. Issued by wave 0 only (TDM ignores EXEC).
    if (tid < 32) {
        const unsigned long long ga =
            (unsigned long long)(const void*)(wfrag + ((size_t)b << 16));
        v4u g0;
        g0[0] = 1u;                                  // count=1, user mode
        g0[1] = 0u;                                  // lds_addr = 0
        g0[2] = (unsigned)(ga & 0xFFFFFFFFull);      // global_addr[31:0]
        g0[3] = (unsigned)((ga >> 32) & 0x1FFFFFFull) | (2u << 30); // [56:32], type=2
        v8i g1;
        g1[0] = (int)(3u << 16);                     // data_size = 8B
        g1[1] = (int)(16384u << 16);                 // tensor_dim0[15:0] = 16384
        g1[2] = (int)(1u << 16);                     // dim0[31:16]=0, tensor_dim1=1
        g1[3] = (int)(16384u << 16);                 // dim1[31:16]=0, tile_dim0=16384
        g1[4] = 1;                                   // tile_dim1=1, tile_dim2=0
        g1[5] = 16384;                               // tensor_dim0_stride[31:0]
        g1[6] = 0;
        g1[7] = 0;
        const v4i gz4 = {};
        const v8i gz8 = {};
        __builtin_amdgcn_tensor_load_to_lds(g0, g1, gz4, gz4, gz8, 0);
        __builtin_amdgcn_s_wait_tensorcnt(0);
    }
    __syncthreads();

    const int wave = tid >> 5;
    const int lane = tid & 31;
    const int half = lane >> 4;
    const int l    = lane & 15;
    const int rowBase = mblk * 128 + wave * 16;

    // A-fragment row this lane reads from
    const float* xrow = x + (((size_t)b * N_ROWS + rowBase + l) * IN_C);

    v8f acc[16] = {};

    #pragma unroll
    for (int kt = 0; kt < 8; ++kt) {
        // A fragment (16x32 bf16):
        //   lane<16 : K = kt*32 + {0..15}
        //   lane>=16: K = kt*32 + {16..31}
        const int k0 = kt * 32 + half * 16;
        const int k1 = k0 + 8;
        const float4 f0 = *(const float4*)(xrow + k0);
        const float4 f1 = *(const float4*)(xrow + k0 + 4);
        const float4 f2 = *(const float4*)(xrow + k1);
        const float4 f3 = *(const float4*)(xrow + k1 + 4);
        v16bf a;
        a[0]  = (__bf16)f0.x; a[1]  = (__bf16)f0.y;
        a[2]  = (__bf16)f0.z; a[3]  = (__bf16)f0.w;
        a[4]  = (__bf16)f1.x; a[5]  = (__bf16)f1.y;
        a[6]  = (__bf16)f1.z; a[7]  = (__bf16)f1.w;
        a[8]  = (__bf16)f2.x; a[9]  = (__bf16)f2.y;
        a[10] = (__bf16)f2.z; a[11] = (__bf16)f2.w;
        a[12] = (__bf16)f3.x; a[13] = (__bf16)f3.y;
        a[14] = (__bf16)f3.z; a[15] = (__bf16)f3.w;

        #pragma unroll
        for (int nt = 0; nt < 16; ++nt) {
            const v16bf bv =
                *(const v16bf*)(smemB + (((kt * 16 + nt) * 32 + lane) << 4));
            acc[nt] = __builtin_amdgcn_wmma_f32_16x16x32_bf16(
                false, a, false, bv, (short)0, acc[nt], false, false);
        }
    }

    // D layout: VGPR r -> row (rowBase + r + half*8), col (nt*16 + l)
    float* ob = out + (((size_t)b * N_ROWS + rowBase + half * 8) * OUT_C) + l;
    #pragma unroll
    for (int nt = 0; nt < 16; ++nt) {
        #pragma unroll
        for (int r = 0; r < 8; ++r) {
            ob[(size_t)r * OUT_C + nt * 16] = acc[nt][r];
        }
    }
}

extern "C" void kernel_launch(void* const* d_in, const int* in_sizes, int n_in,
                              void* d_out, int out_size, void* d_ws, size_t ws_size,
                              hipStream_t stream) {
    const float* x      = (const float*)d_in[0];  // 4*65536*256
    const float* style  = (const float*)d_in[1];  // 4*512
    const float* weight = (const float*)d_in[2];  // 256*256
    const float* mod_w  = (const float*)d_in[3];  // 256*512
    const float* mod_b  = (const float*)d_in[4];  // 256
    float*       out    = (float*)d_out;
    __bf16*      wfrag  = (__bf16*)d_ws;          // needs 512KB

    prep_weights_kernel<<<BATCH, 256, 0, stream>>>(style, weight, mod_w, mod_b, wfrag);
    // 4 batches * 512 row-blocks; 128KB dynamic LDS for the B panel
    modconv_gemm_kernel<<<BATCH * 512, 256, 131072, stream>>>(x, wfrag, out);
}
